// BaseModel_14912126451950
// MI455X (gfx1250) — compile-verified, matrix operand order checked by reading, not run on hardware
//
#include <hip/hip_runtime.h>
#include <hip/hip_bf16.h>
#include <math.h>
#include <stdint.h>

// ---- problem constants (match reference) ----
#define B_    64
#define S_    256
#define D_    512
#define H_    8
#define DH_   64
#define DFF_  2048
#define L_    6
#define NN_   48
#define NE_   96
#define KP_   4
#define NNODE_ 44
#define NEDGE_ 26

typedef __attribute__((ext_vector_type(16))) _Float16 v16h;
typedef __attribute__((ext_vector_type(8)))  _Float16 v8h;
typedef __attribute__((ext_vector_type(8)))  float    v8f;
typedef __attribute__((ext_vector_type(4)))  int      v4i;

// CDNA5 WMMA f32 = f16 x f16 + f32, 16x16x32, wave32.
__device__ __forceinline__ v8f wmma16(v16h a, v16h b, v8f c) {
  return __builtin_amdgcn_wmma_f32_16x16x32_f16(
      /*neg_a=*/false, a, /*neg_b=*/false, b,
      /*c_mod=*/(short)0, c, /*reuse_a=*/false, /*reuse_b=*/false);
}

// ---- CDNA5 async global->LDS copies (ASYNCcnt-tracked), with sync fallback ----
#if __has_builtin(__builtin_amdgcn_global_load_async_to_lds_b128) && \
    __has_builtin(__builtin_amdgcn_s_wait_asynccnt)
#define ASYNC_OK 1
#else
#define ASYNC_OK 0
#endif

__device__ __forceinline__ void async_b128(const void* g, void* l) {
#if ASYNC_OK
  typedef __attribute__((address_space(1))) v4i* gp_t;   // global v4i*
  typedef __attribute__((address_space(3))) v4i* lp_t;   // LDS v4i*
  __builtin_amdgcn_global_load_async_to_lds_b128(
      (gp_t)(uintptr_t)g,
      (lp_t)(unsigned)(uintptr_t)l,
      0, 0);
#else
  *(v8h*)l = *(const v8h*)g;
#endif
}

__device__ __forceinline__ void async_wait0() {
#if ASYNC_OK
  __builtin_amdgcn_s_wait_asynccnt(0);
#endif
}

// Load one 16x32 (rows x K) operand fragment from LDS.
// Per-lane pattern: lanes 0..15 hold K[0..15] of row (lane), lanes 16..31 hold K[16..31].
__device__ __forceinline__ v16h ldsfrag(const _Float16* base, int stride,
                                        int row0, int kofs, int lane) {
  const _Float16* p = base + (size_t)(row0 + (lane & 15)) * stride + kofs + ((lane >> 4) << 4);
  union { v16h v; v8h h[2]; } u;
  u.h[0] = *(const v8h*)p;
  u.h[1] = *(const v8h*)(p + 8);
  return u.v;
}

// ---------------------------------------------------------------------------
// Generic WMMA GEMM: C[M,N] = A[M,K](f16) * W[N,K]^T(f16) + bias[N]
// Block = 256 threads (8 waves), tile 128(M) x 64(N), K-step 32.
// Double-buffered LDS fed by async global->LDS DMA; 4 WMMA per wave per step.
// ---------------------------------------------------------------------------
template <bool OUT16, bool RELU>
__global__ __launch_bounds__(256)
void gemm_bias(const _Float16* __restrict__ A, const _Float16* __restrict__ W,
               const float* __restrict__ bias, float* __restrict__ C32,
               _Float16* __restrict__ C16, int M, int N, int K) {
  __shared__ _Float16 As[2][128 * 40];   // 2 x (128 rows x 32 K), stride 40 (pad)
  __shared__ _Float16 Ws[2][64 * 40];    // 2 x (64 rows x 32 K)

  const int tid  = threadIdx.x;
  const int lane = tid & 31;
  const int wave = tid >> 5;
  const int wm   = wave >> 1;          // 0..3
  const int wn   = wave & 1;           // 0..1
  const int m0   = blockIdx.y * 128;
  const int n0   = blockIdx.x * 64;

  const int ar = tid >> 2, ag = tid & 3;   // staging: rows ar, ar+64; 8-half group ag

  v8f acc[2][2] = {};

  auto stage = [&](int k0, int buf) {
    async_b128(&A[(size_t)(m0 + ar) * K + k0 + ag * 8],      &As[buf][ar * 40 + ag * 8]);
    async_b128(&A[(size_t)(m0 + ar + 64) * K + k0 + ag * 8], &As[buf][(ar + 64) * 40 + ag * 8]);
    async_b128(&W[(size_t)(n0 + ar) * K + k0 + ag * 8],      &Ws[buf][ar * 40 + ag * 8]);
  };

  stage(0, 0);
  async_wait0();
  __syncthreads();

  int buf = 0;
  for (int k0 = 0; k0 < K; k0 += 32) {
    if (k0 + 32 < K) stage(k0 + 32, buf ^ 1);   // prefetch next tile into other buffer

    v16h a0 = ldsfrag(As[buf], 40, wm * 32,      0, lane);
    v16h a1 = ldsfrag(As[buf], 40, wm * 32 + 16, 0, lane);
    v16h b0 = ldsfrag(Ws[buf], 40, wn * 32,      0, lane);
    v16h b1 = ldsfrag(Ws[buf], 40, wn * 32 + 16, 0, lane);
    acc[0][0] = wmma16(a0, b0, acc[0][0]);
    acc[0][1] = wmma16(a0, b1, acc[0][1]);
    acc[1][0] = wmma16(a1, b0, acc[1][0]);
    acc[1][1] = wmma16(a1, b1, acc[1][1]);

    async_wait0();
    __syncthreads();
    buf ^= 1;
  }

  // Epilogue. C frag layout: VGPR i, lanes 0-15 -> row i, lanes 16-31 -> row i+8; col = lane%16.
#pragma unroll
  for (int fi = 0; fi < 2; ++fi)
#pragma unroll
    for (int fj = 0; fj < 2; ++fj) {
#pragma unroll
      for (int i = 0; i < 8; ++i) {
        int row = m0 + wm * 32 + fi * 16 + ((lane >> 4) << 3) + i;
        int col = n0 + wn * 32 + fj * 16 + (lane & 15);
        float v = acc[fi][fj][i] + bias[col];
        if (RELU) v = v > 0.f ? v : 0.f;
        if (OUT16) C16[(size_t)row * N + col] = (_Float16)v;
        else       C32[(size_t)row * N + col] = v;
      }
    }
}

// ---------------------------------------------------------------------------
// Attention: one block per (b, h, 64-row query tile). qkv is f16 [B,S,3D].
// LDS: Q 64x64, K 256x64, V^T 64x256, scores f32 64x256, P f16 64x256.
// ---------------------------------------------------------------------------
__global__ __launch_bounds__(256)
void attention(const _Float16* __restrict__ qkv, _Float16* __restrict__ o) {
  extern __shared__ char smem[];
  _Float16* Qs = (_Float16*)smem;            // 64 x 72
  _Float16* Ks = Qs + 64 * 72;               // 256 x 72
  _Float16* Vt = Ks + 256 * 72;              // 64 x 264  (dh-major)
  float*    Sf = (float*)(Vt + 64 * 264);    // 64 x 264 f32 scores
  _Float16* Pb = (_Float16*)(Sf + 64 * 264); // 64 x 264 f16 probs

  const int tid = threadIdx.x, lane = tid & 31, wave = tid >> 5;
  const int blk = blockIdx.x;
  const int qt = blk & 3;                    // S/64 = 4 query tiles
  const int h  = (blk >> 2) & 7;
  const int b  = blk >> 5;
  const int q0 = qt * 64;
  const size_t rs = 3 * D_;                  // qkv row stride (halves)
  const _Float16* qb = qkv + (size_t)b * S_ * rs + h * DH_;
  const _Float16* kb = qb + D_;
  const _Float16* vb = qb + 2 * D_;

  // stage Q (64 x 64) + K (256 x 64) via async DMA into LDS
  for (int i = tid; i < 64 * 8; i += 256) {
    int r = i >> 3, g = i & 7;
    async_b128(&qb[(size_t)(q0 + r) * rs + g * 8], &Qs[r * 72 + g * 8]);
  }
  for (int i = tid; i < 256 * 8; i += 256) {
    int r = i >> 3, g = i & 7;
    async_b128(&kb[(size_t)r * rs + g * 8], &Ks[r * 72 + g * 8]);
  }
  // stage V transposed: Vt[dh][s] (scalar transpose)
  for (int i = tid; i < 64 * 256; i += 256) {
    int s = i & 255, dh = i >> 8;
    Vt[dh * 264 + s] = vb[(size_t)s * rs + dh];
  }
  async_wait0();
  __syncthreads();

  // scores = Q * K^T * (1/sqrt(64)) : 4x16 tiles of 16x16, K-dim 64 = 2 WMMA
  for (int t = wave; t < 64; t += 8) {
    int ti = t >> 4, tj = t & 15;
    v8f acc = {};
    acc = wmma16(ldsfrag(Qs, 72, ti * 16, 0,  lane),
                 ldsfrag(Ks, 72, tj * 16, 0,  lane), acc);
    acc = wmma16(ldsfrag(Qs, 72, ti * 16, 32, lane),
                 ldsfrag(Ks, 72, tj * 16, 32, lane), acc);
#pragma unroll
    for (int i = 0; i < 8; ++i) {
      int row = ti * 16 + ((lane >> 4) << 3) + i;
      int col = tj * 16 + (lane & 15);
      Sf[row * 264 + col] = acc[i] * 0.125f;
    }
  }
  __syncthreads();

  // row-wise softmax (8 rows per wave, 8 cols per lane), write f16 probs
  for (int r = wave * 8; r < wave * 8 + 8; ++r) {
    float vals[8];
    float mx = -1e30f;
#pragma unroll
    for (int j = 0; j < 8; ++j) {
      vals[j] = Sf[r * 264 + lane + j * 32];
      mx = fmaxf(mx, vals[j]);
    }
#pragma unroll
    for (int off = 16; off > 0; off >>= 1) mx = fmaxf(mx, __shfl_xor(mx, off, 32));
    float sum = 0.f;
#pragma unroll
    for (int j = 0; j < 8; ++j) { vals[j] = __expf(vals[j] - mx); sum += vals[j]; }
#pragma unroll
    for (int off = 16; off > 0; off >>= 1) sum += __shfl_xor(sum, off, 32);
    float inv = 1.f / sum;
#pragma unroll
    for (int j = 0; j < 8; ++j) Pb[r * 264 + lane + j * 32] = (_Float16)(vals[j] * inv);
  }
  __syncthreads();

  // O = P(64x256) * V(256x64): 4x4 tiles, K loop 8 steps of 32
  for (int t = wave; t < 16; t += 8) {
    int ti = t >> 2, tj = t & 3;
    v8f acc = {};
    for (int k0 = 0; k0 < 256; k0 += 32) {
      acc = wmma16(ldsfrag(Pb, 264, ti * 16, k0, lane),
                   ldsfrag(Vt, 264, tj * 16, k0, lane), acc);
    }
#pragma unroll
    for (int i = 0; i < 8; ++i) {
      int row = q0 + ti * 16 + ((lane >> 4) << 3) + i;
      int col = tj * 16 + (lane & 15);
      o[((size_t)(b * S_ + row)) * D_ + h * DH_ + col] = (_Float16)acc[i];
    }
  }
}

// ---------------------------------------------------------------------------
// Residual add + LayerNorm over D=512; writes fp32 h and fp16 mirror.
// ---------------------------------------------------------------------------
__global__ __launch_bounds__(256)
void add_layernorm(const float* __restrict__ res, const float* __restrict__ x,
                   const float* __restrict__ g, const float* __restrict__ bta,
                   float* __restrict__ h32, _Float16* __restrict__ h16) {
  __shared__ float red[8];
  const int row = blockIdx.x, t = threadIdx.x;
  const size_t base = (size_t)row * D_;
  float x0 = res[base + t] + x[base + t];
  float x1 = res[base + t + 256] + x[base + t + 256];

  float s = x0 + x1;
  for (int off = 16; off > 0; off >>= 1) s += __shfl_xor(s, off, 32);
  if ((t & 31) == 0) red[t >> 5] = s;
  __syncthreads();
  float mu = 0.f;
#pragma unroll
  for (int i = 0; i < 8; ++i) mu += red[i];
  mu *= (1.0f / D_);
  __syncthreads();

  float d0 = x0 - mu, d1 = x1 - mu;
  float vs = d0 * d0 + d1 * d1;
  for (int off = 16; off > 0; off >>= 1) vs += __shfl_xor(vs, off, 32);
  if ((t & 31) == 0) red[t >> 5] = vs;
  __syncthreads();
  float var = 0.f;
#pragma unroll
  for (int i = 0; i < 8; ++i) var += red[i];
  var *= (1.0f / D_);
  float rstd = rsqrtf(var + 1e-5f);

  float y0 = d0 * rstd * g[t] + bta[t];
  float y1 = d1 * rstd * g[t + 256] + bta[t + 256];
  h32[base + t] = y0;           h32[base + t + 256] = y1;
  h16[base + t] = (_Float16)y0; h16[base + t + 256] = (_Float16)y1;
}

// embedding * sqrt(S) + sinusoidal positional encoding
__global__ __launch_bounds__(256)
void embed_pos(const int* __restrict__ ids, const float* __restrict__ emb,
               float* __restrict__ h32, _Float16* __restrict__ h16) {
  const int bs = blockIdx.x, t = threadIdx.x;
  const int s = bs % S_;
  const int id = ids[bs];
  const size_t base = (size_t)bs * D_;
#pragma unroll
  for (int r = 0; r < 2; ++r) {
    int d = t + r * 256;
    int i = d >> 1;
    float div = __expf(-(float)(2 * i) * (9.210340371976184f / (float)D_)); // ln(10000)/D
    float ang = (float)s * div;
    float pe = (d & 1) ? __cosf(ang) : __sinf(ang);
    float v = emb[(size_t)id * D_ + d] * 16.0f + pe;  // sqrt(256) = 16
    h32[base + d] = v;
    h16[base + d] = (_Float16)v;
  }
}

__global__ __launch_bounds__(256)
void f32_to_f16(const float* __restrict__ x, _Float16* __restrict__ y, size_t n) {
  size_t i = (size_t)blockIdx.x * blockDim.x + threadIdx.x;
  size_t stride = (size_t)gridDim.x * blockDim.x;
  for (; i < n; i += stride) y[i] = (_Float16)x[i];
}

// node head: mean-pool 4 tokens -> tanh(mean @ Wn^T + bn) * 5
__global__ __launch_bounds__(256)
void node_head(const float* __restrict__ h32, const int* __restrict__ node_ids,
               const float* __restrict__ Wn, const float* __restrict__ bn,
               float* __restrict__ out) {
  __shared__ float mean[D_];
  const int blk = blockIdx.x, t = threadIdx.x;
  const int b = blk / NN_, n = blk % NN_;
  int idx[KP_];
#pragma unroll
  for (int k = 0; k < KP_; ++k) idx[k] = node_ids[((size_t)(b * NN_ + n)) * KP_ + k];
  for (int d = t; d < D_; d += 256) {
    float s = 0.f;
#pragma unroll
    for (int k = 0; k < KP_; ++k) s += h32[((size_t)(b * S_ + idx[k])) * D_ + d];
    mean[d] = s * 0.25f;
  }
  __syncthreads();
  if (t < NNODE_) {
    float acc = bn[t];
    for (int d = 0; d < D_; ++d) acc += mean[d] * Wn[(size_t)t * D_ + d];
    out[((size_t)(b * NN_ + n)) * NNODE_ + t] = tanhf(acc) * 5.0f;
  }
}

// edge head: mean-pool each endpoint group, concat -> tanh(feat @ We^T + be) * 5
__global__ __launch_bounds__(256)
void edge_head(const float* __restrict__ h32, const int* __restrict__ edge_ids,
               const float* __restrict__ We, const float* __restrict__ beb,
               float* __restrict__ out) {
  __shared__ float mean[2 * D_];
  const int blk = blockIdx.x, t = threadIdx.x;
  const int b = blk / NE_, e = blk % NE_;
  for (int d = t; d < 2 * D_; d += 256) {
    int j = d >> 9, dd = d & (D_ - 1);
    float s = 0.f;
#pragma unroll
    for (int k = 0; k < KP_; ++k) {
      int idx = edge_ids[(((size_t)(b * NE_ + e)) * 2 + j) * KP_ + k];
      s += h32[((size_t)(b * S_ + idx)) * D_ + dd];
    }
    mean[d] = s * 0.25f;
  }
  __syncthreads();
  if (t < NEDGE_) {
    float acc = beb[t];
    for (int d = 0; d < 2 * D_; ++d) acc += mean[d] * We[(size_t)t * 2 * D_ + d];
    out[((size_t)(b * NE_ + e)) * NEDGE_ + t] = tanhf(acc) * 5.0f;
  }
}

// ---------------------------------------------------------------------------
extern "C" void kernel_launch(void* const* d_in, const int* in_sizes, int n_in,
                              void* d_out, int out_size, void* d_ws, size_t ws_size,
                              hipStream_t stream) {
  const int*   input_ids = (const int*)d_in[0];
  const int*   node_ids  = (const int*)d_in[1];
  const int*   edge_ids  = (const int*)d_in[2];
  const float* emb   = (const float*)d_in[3];
  const float* Wqkv  = (const float*)d_in[4];
  const float* bqkv  = (const float*)d_in[5];
  const float* Wo    = (const float*)d_in[6];
  const float* bo    = (const float*)d_in[7];
  const float* ln1_g = (const float*)d_in[8];
  const float* ln1_b = (const float*)d_in[9];
  const float* W1    = (const float*)d_in[10];
  const float* b1    = (const float*)d_in[11];
  const float* W2    = (const float*)d_in[12];
  const float* b2    = (const float*)d_in[13];
  const float* ln2_g = (const float*)d_in[14];
  const float* ln2_b = (const float*)d_in[15];
  const float* Wn    = (const float*)d_in[16];
  const float* bn    = (const float*)d_in[17];
  const float* We    = (const float*)d_in[18];
  const float* be    = (const float*)d_in[19];
  float* out = (float*)d_out;

  char* ws = (char*)d_ws;
  size_t off = 0;
  auto carve = [&](size_t bytes) -> char* {
    char* p = ws + off;
    off += (bytes + 255) & ~(size_t)255;
    return p;
  };
  const size_t MR = (size_t)B_ * S_;   // 16384 rows
  float*    h32    = (float*)carve(MR * D_ * 4);
  _Float16* h16    = (_Float16*)carve(MR * D_ * 2);
  _Float16* qkv16  = (_Float16*)carve(MR * 3 * D_ * 2);
  _Float16* att16  = (_Float16*)carve(MR * D_ * 2);
  _Float16* ffn16  = (_Float16*)carve(MR * DFF_ * 2);
  float*    tmp32  = (float*)carve(MR * D_ * 4);
  _Float16* wqkv16 = (_Float16*)carve((size_t)L_ * 3 * D_ * D_ * 2);
  _Float16* wo16   = (_Float16*)carve((size_t)L_ * D_ * D_ * 2);
  _Float16* w116   = (_Float16*)carve((size_t)L_ * DFF_ * D_ * 2);
  _Float16* w216   = (_Float16*)carve((size_t)L_ * D_ * DFF_ * 2);

  auto conv = [&](const float* src, _Float16* dst, size_t n) {
    int blocks = (int)((n + 256 * 8 - 1) / (256 * 8));
    f32_to_f16<<<blocks, 256, 0, stream>>>(src, dst, n);
  };
  conv(Wqkv, wqkv16, (size_t)L_ * 3 * D_ * D_);
  conv(Wo,   wo16,   (size_t)L_ * D_ * D_);
  conv(W1,   w116,   (size_t)L_ * DFF_ * D_);
  conv(W2,   w216,   (size_t)L_ * D_ * DFF_);

  embed_pos<<<B_ * S_, 256, 0, stream>>>(input_ids, emb, h32, h16);

  const int attn_lds = (64*72 + 256*72 + 64*264) * 2 + 64*264 * 4 + 64*264 * 2; // 181248 B

  for (int l = 0; l < L_; ++l) {
    // qkv = h @ Wqkv^T + bqkv   -> f16
    gemm_bias<true, false><<<dim3((3 * D_) / 64, MR / 128), 256, 0, stream>>>(
        h16, wqkv16 + (size_t)l * 3 * D_ * D_, bqkv + (size_t)l * 3 * D_,
        nullptr, qkv16, (int)MR, 3 * D_, D_);
    // self-attention -> att16
    attention<<<B_ * H_ * (S_ / 64), 256, attn_lds, stream>>>(qkv16, att16);
    // o = att @ Wo^T + bo       -> f32
    gemm_bias<false, false><<<dim3(D_ / 64, MR / 128), 256, 0, stream>>>(
        att16, wo16 + (size_t)l * D_ * D_, bo + (size_t)l * D_,
        tmp32, nullptr, (int)MR, D_, D_);
    // h = LN(h + o)
    add_layernorm<<<MR, 256, 0, stream>>>(h32, tmp32,
        ln1_g + (size_t)l * D_, ln1_b + (size_t)l * D_, h32, h16);
    // f = relu(h @ W1^T + b1)   -> f16
    gemm_bias<true, true><<<dim3(DFF_ / 64, MR / 128), 256, 0, stream>>>(
        h16, w116 + (size_t)l * DFF_ * D_, b1 + (size_t)l * DFF_,
        nullptr, ffn16, (int)MR, DFF_, D_);
    // f2 = f @ W2^T + b2        -> f32
    gemm_bias<false, false><<<dim3(D_ / 64, MR / 128), 256, 0, stream>>>(
        ffn16, w216 + (size_t)l * D_ * DFF_, b2 + (size_t)l * D_,
        tmp32, nullptr, (int)MR, D_, DFF_);
    // h = LN(h + f2)
    add_layernorm<<<MR, 256, 0, stream>>>(h32, tmp32,
        ln2_g + (size_t)l * D_, ln2_b + (size_t)l * D_, h32, h16);
  }

  node_head<<<B_ * NN_, 256, 0, stream>>>(h32, node_ids, Wn, bn, out);
  edge_head<<<B_ * NE_, 256, 0, stream>>>(h32, edge_ids, We, be,
                                          out + (size_t)B_ * NN_ * NNODE_);
}
